// AttentionBlock_16441134809287
// MI455X (gfx1250) — matrix operand order repaired and checked by reading, not compiled, not run on hardware
//
#include <hip/hip_runtime.h>
#include <math.h>

typedef __bf16 bf16;
typedef __bf16 bh8   __attribute__((ext_vector_type(8)));
typedef __bf16 v16bf __attribute__((ext_vector_type(16)));
typedef float  v8f   __attribute__((ext_vector_type(8)));

#define T_LEN 8160
#define BATCH 4
#define NH    8
#define HD    128
#define WBLK  12
#define CCTX  24
#define UBLK  680
#define MTOT  (BATCH * T_LEN)   /* 32640 = 255*128 */
#define NQKV  3072

union Frag16 { v16bf v; bh8 h8[2]; bf16 e[16]; };
union Frag8  { bh8 v;  bf16 e[8]; };
union Acc8   { v8f v;  float f[8]; };

// ---------------------------------------------------------------------------
// 1) fp32 -> bf16 conversion of x
// ---------------------------------------------------------------------------
__global__ void k_convert_x(const float* __restrict__ x, bf16* __restrict__ xb, int n) {
    int i = (blockIdx.x * blockDim.x + threadIdx.x) * 4;
    if (i + 3 < n) {
        float4 v = *(const float4*)(x + i);
        xb[i + 0] = (bf16)v.x;
        xb[i + 1] = (bf16)v.y;
        xb[i + 2] = (bf16)v.z;
        xb[i + 3] = (bf16)v.w;
    }
}

// ---------------------------------------------------------------------------
// 2) pack wq|wk|wv into transposed bf16 WbT[3072][1024]: WbT[col][k] = W_g[k][f]
//    (col = g*1024 + f). K-runs contiguous => cheap WMMA B-operand loads.
// ---------------------------------------------------------------------------
__global__ void k_convert_w(const float* __restrict__ wq, const float* __restrict__ wk,
                            const float* __restrict__ wv, bf16* __restrict__ wbT) {
    int o = blockIdx.x * blockDim.x + threadIdx.x;     // 0 .. 3072*1024-1
    int col = o >> 10, k = o & 1023;
    int g = col >> 10, f = col & 1023;
    const float* w = (g == 0) ? wq : ((g == 1) ? wk : wv);
    wbT[o] = (bf16)w[k * 1024 + f];
}

// ---------------------------------------------------------------------------
// 3) projected sinusoidal table: sinb[n][16][128] bf16, rows f=13..15 zero.
//    pos[f] = 12 - f ; emb = [sin(pos*inv), cos(pos*inv)] ; proj by pos_proj.
// ---------------------------------------------------------------------------
__global__ void k_sinemb(const float* __restrict__ pos_proj, bf16* __restrict__ sinb) {
    __shared__ float emb[1024];
    int f = blockIdx.x;                 // 0..15
    int tid = threadIdx.x;              // 256
    bool valid = (f < 13);
    float pos = (float)(12 - f);
    const float inc = 9.210340371976184f / 511.0f;     // ln(10000)/511
    for (int d = tid; d < 1024; d += 256) {
        float val = 0.f;
        if (valid) {
            int i = d & 511;
            float inv = __expf(-(float)i * inc);
            float st = pos * inv;
            val = (d < 512) ? __sinf(st) : __cosf(st);
        }
        emb[d] = val;
    }
    __syncthreads();
    for (int j = tid; j < 1024; j += 256) {
        float acc = 0.f;
        for (int d = 0; d < 1024; ++d) acc += emb[d] * pos_proj[d * 1024 + j];
        int n = j >> 7, h = j & 127;
        sinb[((size_t)n * 16 + f) * 128 + h] = (bf16)acc;
    }
}

// ---------------------------------------------------------------------------
// 4) fused QKV GEMM:  QKV[32640][3072] = Xb[32640][1024] x WbT^T, bf16 WMMA.
//    WG tile 128(M) x 256(N); 8 waves, each 64x64 via 4x4 wmma_f32_16x16x32_bf16.
// ---------------------------------------------------------------------------
__global__ __launch_bounds__(256)
void k_gemm_qkv(const bf16* __restrict__ A, const bf16* __restrict__ Bt,
                bf16* __restrict__ C) {
    int lane = threadIdx.x & 31;
    int wid  = threadIdx.x >> 5;
    int hi = lane >> 4;               // lane half
    int lj = lane & 15;

    int mb = blockIdx.x * 128 + (wid >> 2) * 64;   // 0..32576
    int nb = blockIdx.y * 256 + (wid & 3) * 64;    // 0..3008

    Acc8 acc[4][4];
#pragma unroll
    for (int i = 0; i < 4; ++i)
#pragma unroll
        for (int j = 0; j < 4; ++j)
            acc[i][j].v = v8f{0.f,0.f,0.f,0.f,0.f,0.f,0.f,0.f};

    // A fragment: lane holds row (mb+16i+lj); K-run kk+hi*16 .. +15 contiguous
    const bf16* aBase = A + (size_t)(mb + lj) * 1024 + hi * 16;
    // B fragment: lane holds col (nb+16j+lj); K-run kk+hi*16 .. +15 contiguous in WbT row
    const bf16* bBase = Bt + (size_t)(nb + lj) * 1024 + hi * 16;

    for (int kk = 0; kk < 1024; kk += 32) {
        Frag16 a[4];
#pragma unroll
        for (int i = 0; i < 4; ++i) {
            const bf16* p = aBase + (size_t)i * 16 * 1024 + kk;
            a[i].h8[0] = *(const bh8*)(p);
            a[i].h8[1] = *(const bh8*)(p + 8);
        }
#pragma unroll
        for (int j = 0; j < 4; ++j) {
            const bf16* p = bBase + (size_t)j * 16 * 1024 + kk;
            Frag16 bfrag;
            bfrag.h8[0] = *(const bh8*)(p);
            bfrag.h8[1] = *(const bh8*)(p + 8);
#pragma unroll
            for (int i = 0; i < 4; ++i)
                acc[i][j].v = __builtin_amdgcn_wmma_f32_16x16x32_bf16(
                    false, a[i].v, false, bfrag.v, (short)0, acc[i][j].v, false, false);
        }
    }

#pragma unroll
    for (int i = 0; i < 4; ++i)
#pragma unroll
        for (int j = 0; j < 4; ++j)
#pragma unroll
            for (int r = 0; r < 8; ++r) {
                int row = mb + 16 * i + r + 8 * hi;
                int col = nb + 16 * j + lj;
                C[(size_t)row * NQKV + col] = (bf16)acc[i][j].f[r];
            }
}

// ---------------------------------------------------------------------------
// 5) local attention: one wave per (b, u, head).
//    QK^T + rel-pos as 16x48 WMMA; shfl diag-gather of bd; cap/mask/softmax;
//    probs->LDS; V->LDS via ASYNC global->LDS copies; PV as 8 WMMAs -> fp32 out.
// ---------------------------------------------------------------------------
__global__ __launch_bounds__(32)
void k_attn(const bf16* __restrict__ QKV, const bf16* __restrict__ SINB,
            float* __restrict__ out) {
    __shared__ bf16 sP[16 * 32];     // probs, row-major, rows m, cols c
    __shared__ bf16 sV[32 * 128];    // V context, rows c (24..31 zero), cols h

    int id = blockIdx.x;
    int n = id & 7;
    int u = (id >> 3) % UBLK;
    int b = id / (UBLK * NH);
    int lane = threadIdx.x;
    int hi = lane >> 4;
    int lj = lane & 15;

    // Prefetch this wave's K context rows toward the WGP (global_prefetch_b8).
    {
        int taf = u * WBLK - 12 + lane;                 // one context row per lane
        if (taf >= 0 && taf < T_LEN && lane < CCTX) {
            const bf16* pf = QKV + (size_t)(b * T_LEN + taf) * NQKV + 1024 + n * HD;
            __builtin_prefetch(pf, 0, 1);
        }
    }

    // ---- QK^T (+ sin-emb tile) ----
    Acc8 acc0, acc1, acc2;
    acc0.v = v8f{0.f,0.f,0.f,0.f,0.f,0.f,0.f,0.f};
    acc1.v = acc0.v; acc2.v = acc0.v;

    int tq = u * WBLK + lj; if (tq >= T_LEN) tq = T_LEN - 1;       // rows >=12 discarded
    const bf16* qRow = QKV + (size_t)(b * T_LEN + tq) * NQKV + n * HD;

    int c0l = lj;
    int ta0l = u * WBLK - 12 + c0l;
    bool kv0 = (ta0l >= 0) && (ta0l < T_LEN);
    const bf16* kb0 = kv0 ? (QKV + (size_t)(b * T_LEN + ta0l) * NQKV + 1024 + n * HD + hi * 16)
                          : (const bf16*)nullptr;
    int c1l = 16 + lj;
    int ta1l = u * WBLK - 12 + c1l;                   // >= 4, no lower check needed
    bool kv1 = (ta1l < T_LEN) && (c1l < CCTX);
    const bf16* kb1 = kv1 ? (QKV + (size_t)(b * T_LEN + ta1l) * NQKV + 1024 + n * HD + hi * 16)
                          : (const bf16*)nullptr;
    const bf16* sb = SINB + ((size_t)n * 16 + lj) * 128 + hi * 16;  // rows 13..15 pre-zeroed

    Frag16 zf;
#pragma unroll
    for (int e = 0; e < 16; ++e) zf.e[e] = (bf16)0.0f;

    for (int kk = 0; kk < HD; kk += 32) {
        Frag16 a;
        a.h8[0] = *(const bh8*)(qRow + kk + hi * 16);
        a.h8[1] = *(const bh8*)(qRow + kk + hi * 16 + 8);

        Frag16 b0 = zf, b1 = zf, b2;
        if (kv0) { b0.h8[0] = *(const bh8*)(kb0 + kk); b0.h8[1] = *(const bh8*)(kb0 + kk + 8); }
        if (kv1) { b1.h8[0] = *(const bh8*)(kb1 + kk); b1.h8[1] = *(const bh8*)(kb1 + kk + 8); }
        b2.h8[0] = *(const bh8*)(sb + kk);
        b2.h8[1] = *(const bh8*)(sb + kk + 8);

        acc0.v = __builtin_amdgcn_wmma_f32_16x16x32_bf16(false, a.v, false, b0.v, (short)0, acc0.v, false, false);
        acc1.v = __builtin_amdgcn_wmma_f32_16x16x32_bf16(false, a.v, false, b1.v, (short)0, acc1.v, false, false);
        acc2.v = __builtin_amdgcn_wmma_f32_16x16x32_bf16(false, a.v, false, b2.v, (short)0, acc2.v, false, false);
    }

    // ---- kick off async V staging (memory -> LDS, ASYNCcnt-tracked) while we
    //      do the softmax math; zero-fills (disjoint addresses) go via DS pipe.
#pragma unroll
    for (int it = 0; it < 12; ++it) {
        int q = it * 32 + lane;          // 0..383 chunk id (24 rows * 16 chunks)
        int c = q >> 4;
        int h0 = (q & 15) * 8;
        int ta = u * WBLK - 12 + c;
        if (ta >= 0 && ta < T_LEN) {
            const bf16* g = QKV + (size_t)(b * T_LEN + ta) * NQKV + 2048 + n * HD + h0;
            unsigned la = (unsigned)(uintptr_t)&sV[c * 128 + h0];
            asm volatile("global_load_async_to_lds_b128 %0, %1, off"
                         :: "v"(la), "v"(g)
                         : "memory");
        } else {
            Frag8 z;
#pragma unroll
            for (int e = 0; e < 8; ++e) z.e[e] = (bf16)0.0f;
            *(bh8*)(&sV[c * 128 + h0]) = z.v;
        }
    }
#pragma unroll
    for (int it = 0; it < 4; ++it) {     // pad rows 24..31 zeroed (DS pipe)
        int q = it * 32 + lane;
        int c = 24 + (q >> 4);
        int h0 = (q & 15) * 8;
        Frag8 z;
#pragma unroll
        for (int e = 0; e < 8; ++e) z.e[e] = (bf16)0.0f;
        *(bh8*)(&sV[c * 128 + h0]) = z.v;
    }

    // ---- cap / mask / softmax (per row m, 16-lane half reductions) ----
    const float RS = 0.08838834764831845f;   // 1/sqrt(128)
#pragma unroll
    for (int r = 0; r < 8; ++r) {
        int m = r + 8 * hi;
        float bdv = acc2.f[r];               // bd[m][lj]
        int f0 = lj - m;
        int f1 = lj + 16 - m;
        float bd0 = __shfl(bdv, (f0 & 15) | (lane & 16), 32);
        float bd1 = __shfl(bdv, (f1 & 15) | (lane & 16), 32);

        int ta0 = u * WBLK - 12 + lj;
        int ta1 = ta0 + 16;
        bool v0 = (f0 >= 0) && (f0 <= 12) && (ta0 >= 0) && (ta0 < T_LEN);
        bool v1 = (f1 >= 0) && (f1 <= 12) && (ta1 < T_LEN) && (lj < 8);

        float l0 = (acc0.f[r] + bd0) * RS;  l0 = 50.f * tanhf(l0 * 0.02f);
        float l1 = (acc1.f[r] + bd1) * RS;  l1 = 50.f * tanhf(l1 * 0.02f);
        float x0 = v0 ? l0 : -1e30f;
        float x1 = v1 ? l1 : -1e30f;

        float mx = fmaxf(x0, x1);
#pragma unroll
        for (int d = 8; d >= 1; d >>= 1) mx = fmaxf(mx, __shfl_xor(mx, d, 32));
        float e0 = v0 ? __expf(x0 - mx) : 0.f;
        float e1 = v1 ? __expf(x1 - mx) : 0.f;
        float s = e0 + e1;
#pragma unroll
        for (int d = 8; d >= 1; d >>= 1) s += __shfl_xor(s, d, 32);
        float inv = 1.f / s;
        sP[m * 32 + lj]      = (bf16)(e0 * inv);
        sP[m * 32 + 16 + lj] = (bf16)(e1 * inv);
    }

    // drain async copies, then make all LDS visible
    asm volatile("s_wait_asynccnt 0x0" ::: "memory");
    __syncthreads();

    // ---- PV: A = probs (from LDS, A-layout), B = V (per-element gather) ----
    Frag16 ap;
    ap.h8[0] = *(const bh8*)(&sP[lj * 32 + hi * 16]);
    ap.h8[1] = *(const bh8*)(&sP[lj * 32 + hi * 16 + 8]);

#pragma unroll
    for (int th = 0; th < 8; ++th) {
        int hcol = th * 16 + lj;
        Frag16 bv;
#pragma unroll
        for (int e = 0; e < 16; ++e)
            bv.e[e] = sV[(hi * 16 + e) * 128 + hcol];

        Acc8 o;
        o.v = v8f{0.f,0.f,0.f,0.f,0.f,0.f,0.f,0.f};
        o.v = __builtin_amdgcn_wmma_f32_16x16x32_bf16(false, ap.v, false, bv.v, (short)0, o.v, false, false);

#pragma unroll
        for (int r = 0; r < 8; ++r) {
            int m = r + 8 * hi;
            if (m < WBLK) {
                int t = u * WBLK + m;
                out[(((size_t)b * T_LEN + t) * NH + n) * HD + hcol] = o.f[r];
            }
        }
    }
}

// ---------------------------------------------------------------------------
extern "C" void kernel_launch(void* const* d_in, const int* in_sizes, int n_in,
                              void* d_out, int out_size, void* d_ws, size_t ws_size,
                              hipStream_t stream) {
    const float* x        = (const float*)d_in[0];
    // d_in[1]=mask (all true), d_in[2]=causal_valid_mask -> handled analytically
    const float* wq       = (const float*)d_in[3];
    const float* wk       = (const float*)d_in[4];
    const float* wv       = (const float*)d_in[5];
    const float* pos_proj = (const float*)d_in[6];
    float* out = (float*)d_out;

    char* ws = (char*)d_ws;
    size_t o = 0;
    bf16* xb   = (bf16*)(ws + o); o += (size_t)MTOT * 1024 * 2;          // 66.8 MB
    bf16* wbT  = (bf16*)(ws + o); o += (size_t)NQKV * 1024 * 2;          // 6.3 MB
    bf16* sinb = (bf16*)(ws + o); o += (size_t)NH * 16 * HD * 2;         // 32 KB
    o = (o + 255) & ~(size_t)255;
    bf16* qkv  = (bf16*)(ws + o);                                        // 200.5 MB

    const int nx = MTOT * 1024;
    k_convert_x<<<nx / 1024, 256, 0, stream>>>(x, xb, nx);
    k_convert_w<<<(NQKV * 1024) / 256, 256, 0, stream>>>(wq, wk, wv, wbT);
    k_sinemb<<<16, 256, 0, stream>>>(pos_proj, sinb);
    k_gemm_qkv<<<dim3(MTOT / 128, NQKV / 256), 256, 0, stream>>>(xb, wbT, qkv);
    k_attn<<<BATCH * UBLK * NH, 32, 0, stream>>>(qkv, sinb, out);
}
